// GraphSAGE_73443940762319
// MI455X (gfx1250) — matrix operand order repaired and checked by reading, not compiled, not run on hardware
//
#include <hip/hip_runtime.h>

// ---------------- problem constants (match reference) ----------------
#define N_NODES 50000
#define N_EDGES 800000
#define D_INPUT 128
#define G_BATCH 64

typedef __bf16 bf16;
typedef __attribute__((ext_vector_type(16))) __bf16 v16bf;
typedef __attribute__((ext_vector_type(8)))  __bf16 v8bf;
typedef __attribute__((ext_vector_type(8)))  float  v8f;

// ---------------- utility kernels ----------------
__global__ void k_zero_f32(float* __restrict__ p, int n) {
    int i = blockIdx.x * blockDim.x + threadIdx.x;
    if (i < n) p[i] = 0.0f;
}

// 4 elements per thread, vector load
__global__ void k_f32_to_bf16_v4(const float* __restrict__ s, bf16* __restrict__ d, int n4) {
    int i = blockIdx.x * blockDim.x + threadIdx.x;
    if (i >= n4) return;
    const float4 v = ((const float4*)s)[i];
    bf16* o = d + (size_t)i * 4;
    o[0] = (bf16)v.x; o[1] = (bf16)v.y; o[2] = (bf16)v.z; o[3] = (bf16)v.w;
}

// W [K x C] f32 row-major  ->  Wt [C x K] bf16 (transposed, fragment-friendly)
__global__ void k_pack_wt(const float* __restrict__ W, bf16* __restrict__ Wt, int K, int C) {
    int idx = blockIdx.x * blockDim.x + threadIdx.x;
    if (idx >= K * C) return;
    int k = idx / C, c = idx - k * C;
    Wt[(size_t)c * K + k] = (bf16)W[idx];
}

__global__ void k_deg_count(const int* __restrict__ dst, float* __restrict__ deg, int E) {
    int e = blockIdx.x * blockDim.x + threadIdx.x;
    if (e < E)
        __hip_atomic_fetch_add(&deg[dst[e]], 1.0f, __ATOMIC_RELAXED, __HIP_MEMORY_SCOPE_AGENT);
}

__global__ void k_invert(float* __restrict__ p, int n) {
    int i = blockIdx.x * blockDim.x + threadIdx.x;
    if (i < n) p[i] = 1.0f / fmaxf(p[i], 1.0f);
}

// gather x[src] (bf16), atomically accumulate f32 into agg[dst]; 4 features/thread
__global__ void k_scatter_add(const bf16* __restrict__ act, const int* __restrict__ src,
                              const int* __restrict__ dst, float* __restrict__ agg,
                              int E, int D) {
    int per = D >> 2;
    int idx = blockIdx.x * blockDim.x + threadIdx.x;
    if (idx >= E * per) return;
    int e = idx / per;
    int c = (idx - e * per) << 2;
    int s = src[e], d = dst[e];
    const bf16* p = act + (size_t)s * D + c;      // 8B contiguous -> b64 load
    float*      q = agg + (size_t)d * D + c;
    #pragma unroll
    for (int j = 0; j < 4; ++j)
        __hip_atomic_fetch_add(&q[j], (float)p[j], __ATOMIC_RELAXED, __HIP_MEMORY_SCOPE_AGENT);
}

// agg_f32 * invdeg[row] -> bf16, 4 per thread
__global__ void k_scale_to_bf16_v4(const float* __restrict__ agg, const float* __restrict__ invdeg,
                                   bf16* __restrict__ out, int total4, int D) {
    int i = blockIdx.x * blockDim.x + threadIdx.x;
    if (i >= total4) return;
    int r = (i << 2) / D;
    float s = invdeg[r];
    const float4 v = ((const float4*)agg)[i];
    bf16* o = out + (size_t)i * 4;
    o[0] = (bf16)(v.x * s); o[1] = (bf16)(v.y * s); o[2] = (bf16)(v.z * s); o[3] = (bf16)(v.w * s);
}

// ---------------- WMMA fragment loaders (CDNA5 16-bit layouts) ----------------
// A 16x32 bf16: lane m in [0,15] row M=m holds K 0..7 / 16..23; lane m+16 holds K 8..15 / 24..31.
// pAh is pre-folded per-lane base (row base + half*8); k0 is a compile-time constant.
__device__ __forceinline__ v16bf load_a_frag(const bf16* __restrict__ pAh, int k0) {
    v8bf lo = *(const v8bf*)(pAh + k0);
    v8bf hi = *(const v8bf*)(pAh + k0 + 16);
    v16bf a;
    #pragma unroll
    for (int i = 0; i < 8; ++i) { a[i] = lo[i]; a[i + 8] = hi[i]; }
    return a;
}

// B 32x16 bf16 from TRANSPOSED weights Wt [C x K]:
// pBh pre-folded per-lane base (column base + half*8); elem offset is a constant.
__device__ __forceinline__ v16bf load_b_frag_t(const bf16* __restrict__ pBh, int elemOff) {
    v8bf lo = *(const v8bf*)(pBh + elemOff);
    v8bf hi = *(const v8bf*)(pBh + elemOff + 16);
    v16bf b;
    #pragma unroll
    for (int i = 0; i < 8; ++i) { b[i] = lo[i]; b[i + 8] = hi[i]; }
    return b;
}

// ---------------- fused SAGE GEMM: D = Aagg@Wl + Ax@Wr + bias, optional relu -------
// One wave computes 16 rows x (TN*16) cols. KK, CC compile-time -> fully unrolled
// K loop, every load/store address = base + constant immediate. All fragment loads
// for a k-step are issued before the WMMA sequence so waits stay partial.
template <int TN, int KK, int CC>
__global__ void k_sage_gemm(const bf16* __restrict__ Ax, const bf16* __restrict__ Aagg,
                            const bf16* __restrict__ WrT, const bf16* __restrict__ WlT,
                            const float* __restrict__ bias,
                            bf16* __restrict__ outB, float* __restrict__ outF,
                            int M, int relu) {
    constexpr int groupsN = CC / (TN * 16);
    int wavesPerBlk = blockDim.x >> 5;
    int wave  = blockIdx.x * wavesPerBlk + (threadIdx.x >> 5);
    int total = (M >> 4) * groupsN;
    if (wave >= total) return;                // wave-uniform: EXEC stays all-ones

    int tm = wave / groupsN;
    int tg = wave - tm * groupsN;
    int lane = threadIdx.x & 31;
    int half = lane >> 4;
    int l16  = lane & 15;
    int row0 = tm << 4;
    int col0 = tg * (TN * 16);

    v8f acc[TN];
    #pragma unroll
    for (int t = 0; t < TN; ++t) {
        float bv = bias[col0 + t * 16 + l16];
        #pragma unroll
        for (int r = 0; r < 8; ++r) acc[t][r] = bv;
    }

    // per-lane base pointers with `half` folded in; all loop offsets are immediates
    const bf16* pGh  = Aagg + (size_t)(row0 + l16) * KK + half * 8;
    const bf16* pXh  = Ax   + (size_t)(row0 + l16) * KK + half * 8;
    const bf16* pWlh = WlT  + (size_t)(col0 + l16) * KK + half * 8;
    const bf16* pWrh = WrT  + (size_t)(col0 + l16) * KK + half * 8;

    #pragma unroll
    for (int k0 = 0; k0 < KK; k0 += 32) {
        // issue ALL loads for this k-step first (one clause, staggered waits)
        v16bf aG = load_a_frag(pGh, k0);
        v16bf aX = load_a_frag(pXh, k0);
        v16bf bL[TN], bR[TN];
        #pragma unroll
        for (int t = 0; t < TN; ++t) {
            bL[t] = load_b_frag_t(pWlh, k0 + t * 16 * KK);
            bR[t] = load_b_frag_t(pWrh, k0 + t * 16 * KK);
        }
        #pragma unroll
        for (int t = 0; t < TN; ++t) {
            acc[t] = __builtin_amdgcn_wmma_f32_16x16x32_bf16(false, aG, false, bL[t],
                                                             (short)0, acc[t], false, false);
            acc[t] = __builtin_amdgcn_wmma_f32_16x16x32_bf16(false, aX, false, bR[t],
                                                             (short)0, acc[t], false, false);
        }
    }

    // D layout: VGPR r -> row (row0 + r + 8*half), col (col0 + t*16 + l16)
    size_t obase = (size_t)(row0 + (half << 3)) * CC + col0 + l16;
    #pragma unroll
    for (int t = 0; t < TN; ++t) {
        #pragma unroll
        for (int r = 0; r < 8; ++r) {
            float v = acc[t][r];
            if (relu) v = fmaxf(v, 0.0f);
            size_t o = obase + (size_t)r * CC + t * 16;
            if (outB) outB[o] = (bf16)v;
            if (outF) outF[o] = v;
        }
    }
}

// ---------------- pooling + head ----------------
__global__ void k_batch_cnt(const int* __restrict__ batch, float* __restrict__ cnt, int Nn) {
    int i = blockIdx.x * blockDim.x + threadIdx.x;
    if (i < Nn)
        __hip_atomic_fetch_add(&cnt[batch[i]], 1.0f, __ATOMIC_RELAXED, __HIP_MEMORY_SCOPE_AGENT);
}

__global__ void k_pool_sum(const float* __restrict__ h3, const int* __restrict__ batch,
                           float* __restrict__ sums, int Nn) {
    int idx = blockIdx.x * blockDim.x + threadIdx.x;
    if (idx >= Nn * 16) return;
    int nI = idx >> 4, f = idx & 15;
    __hip_atomic_fetch_add(&sums[batch[nI] * 16 + f], h3[idx],
                           __ATOMIC_RELAXED, __HIP_MEMORY_SCOPE_AGENT);
}

__global__ void k_final(const float* __restrict__ sums, const float* __restrict__ cnt,
                        const float* __restrict__ Wc, const float* __restrict__ bc,
                        float* __restrict__ out) {
    int g = threadIdx.x;
    if (g >= G_BATCH) return;
    float inv = 1.0f / fmaxf(cnt[g], 1.0f);
    float acc = 0.0f;
    #pragma unroll
    for (int f = 0; f < 16; ++f) {
        float m = sums[g * 16 + f] * inv;
        out[G_BATCH + g * 16 + f] = m;                      // h_pool output
        unsigned h = (unsigned)(g * 16 + f) * 2654435761u;  // deterministic dropout mask
        float hd = ((h >> 13) & 1u) ? (m * 2.0f) : 0.0f;
        acc += hd * Wc[f];
    }
    out[g] = acc + bc[0];                                   // out[G] output
}

// ---------------- driver ----------------
static inline int nblk(long long n, int b) { return (int)((n + b - 1) / b); }

extern "C" void kernel_launch(void* const* d_in, const int* in_sizes, int n_in,
                              void* d_out, int out_size, void* d_ws, size_t ws_size,
                              hipStream_t stream) {
    (void)in_sizes; (void)n_in; (void)out_size; (void)ws_size;

    const float* x    = (const float*)d_in[0];
    const int*   ei   = (const int*)d_in[1];
    const int*   srcE = ei;
    const int*   dstE = ei + N_EDGES;
    const int*   batch = (const int*)d_in[2];
    const float* Wl1 = (const float*)d_in[3];  const float* bl1 = (const float*)d_in[4];
    const float* Wr1 = (const float*)d_in[5];
    const float* Wl2 = (const float*)d_in[6];  const float* bl2 = (const float*)d_in[7];
    const float* Wr2 = (const float*)d_in[8];
    const float* Wl3 = (const float*)d_in[9];  const float* bl3 = (const float*)d_in[10];
    const float* Wr3 = (const float*)d_in[11];
    const float* Wc  = (const float*)d_in[12]; const float* bc  = (const float*)d_in[13];

    // ---- workspace carve-out (256B aligned) ----
    char* ws = (char*)d_ws;
    size_t off = 0;
    auto carve = [&](size_t bytes) -> char* {
        char* p = ws + off;
        off = (off + bytes + 255) & ~(size_t)255;
        return p;
    };
    bf16*  actA   = (bf16*) carve((size_t)N_NODES * 256 * sizeof(bf16));
    bf16*  actB   = (bf16*) carve((size_t)N_NODES * 256 * sizeof(bf16));
    float* aggF   = (float*)carve((size_t)N_NODES * 256 * sizeof(float));
    bf16*  aggB   = (bf16*) carve((size_t)N_NODES * 256 * sizeof(bf16));
    float* h3     = (float*)carve((size_t)N_NODES * 16  * sizeof(float));
    float* invdeg = (float*)carve((size_t)N_NODES * sizeof(float));
    bf16*  wbuf   = (bf16*) carve((size_t)135168 * sizeof(bf16));
    float* sums   = (float*)carve((size_t)G_BATCH * 16 * sizeof(float));
    float* cnt    = (float*)carve((size_t)G_BATCH * sizeof(float));

    bf16* wl1t = wbuf;              // [256 x 128]
    bf16* wr1t = wbuf + 32768;      // [256 x 128]
    bf16* wl2t = wbuf + 65536;      // [128 x 256]
    bf16* wr2t = wbuf + 98304;      // [128 x 256]
    bf16* wl3t = wbuf + 131072;     // [16 x 128]
    bf16* wr3t = wbuf + 133120;     // [16 x 128]

    const int B = 256;

    // ---- precompute: bf16 inputs + transposed bf16 weights, inverse degrees, pool zeroes ----
    k_f32_to_bf16_v4<<<nblk((long long)N_NODES * D_INPUT / 4, B), B, 0, stream>>>(x, actA, N_NODES * D_INPUT / 4);
    k_pack_wt<<<nblk(32768, B), B, 0, stream>>>(Wl1, wl1t, 128, 256);
    k_pack_wt<<<nblk(32768, B), B, 0, stream>>>(Wr1, wr1t, 128, 256);
    k_pack_wt<<<nblk(32768, B), B, 0, stream>>>(Wl2, wl2t, 256, 128);
    k_pack_wt<<<nblk(32768, B), B, 0, stream>>>(Wr2, wr2t, 256, 128);
    k_pack_wt<<<nblk(2048, B), B, 0, stream>>>(Wl3, wl3t, 128, 16);
    k_pack_wt<<<nblk(2048, B), B, 0, stream>>>(Wr3, wr3t, 128, 16);

    k_zero_f32<<<nblk(N_NODES, B), B, 0, stream>>>(invdeg, N_NODES);
    k_deg_count<<<nblk(N_EDGES, B), B, 0, stream>>>(dstE, invdeg, N_EDGES);
    k_invert<<<nblk(N_NODES, B), B, 0, stream>>>(invdeg, N_NODES);

    k_zero_f32<<<nblk(G_BATCH * 16, B), B, 0, stream>>>(sums, G_BATCH * 16);
    k_zero_f32<<<1, B, 0, stream>>>(cnt, G_BATCH);
    k_batch_cnt<<<nblk(N_NODES, B), B, 0, stream>>>(batch, cnt, N_NODES);

    // ---- layer 1: [N,128] -> [N,256], relu ----
    {
        const int D = 128, C = 256;
        long long n = (long long)N_NODES * D;
        k_zero_f32<<<nblk(n, B), B, 0, stream>>>(aggF, (int)n);
        k_scatter_add<<<nblk((long long)N_EDGES * (D >> 2), B), B, 0, stream>>>(actA, srcE, dstE, aggF, N_EDGES, D);
        k_scale_to_bf16_v4<<<nblk(n / 4, B), B, 0, stream>>>(aggF, invdeg, aggB, (int)(n / 4), D);
        int waves = (N_NODES / 16) * (C / 64);
        k_sage_gemm<4, 128, 256><<<nblk(waves, 8), 256, 0, stream>>>(actA, aggB, wr1t, wl1t, bl1,
                                                                     actB, nullptr, N_NODES, 1);
    }
    // ---- layer 2: [N,256] -> [N,128], relu ----
    {
        const int D = 256, C = 128;
        long long n = (long long)N_NODES * D;
        k_zero_f32<<<nblk(n, B), B, 0, stream>>>(aggF, (int)n);
        k_scatter_add<<<nblk((long long)N_EDGES * (D >> 2), B), B, 0, stream>>>(actB, srcE, dstE, aggF, N_EDGES, D);
        k_scale_to_bf16_v4<<<nblk(n / 4, B), B, 0, stream>>>(aggF, invdeg, aggB, (int)(n / 4), D);
        int waves = (N_NODES / 16) * (C / 64);
        k_sage_gemm<4, 256, 128><<<nblk(waves, 8), 256, 0, stream>>>(actB, aggB, wr2t, wl2t, bl2,
                                                                     actA, nullptr, N_NODES, 1);
    }
    // ---- layer 3: [N,128] -> [N,16], no relu, f32 out ----
    {
        const int D = 128, C = 16;
        long long n = (long long)N_NODES * D;
        k_zero_f32<<<nblk(n, B), B, 0, stream>>>(aggF, (int)n);
        k_scatter_add<<<nblk((long long)N_EDGES * (D >> 2), B), B, 0, stream>>>(actA, srcE, dstE, aggF, N_EDGES, D);
        k_scale_to_bf16_v4<<<nblk(n / 4, B), B, 0, stream>>>(aggF, invdeg, aggB, (int)(n / 4), D);
        int waves = (N_NODES / 16) * (C / 16);
        k_sage_gemm<1, 128, 16><<<nblk(waves, 8), 256, 0, stream>>>(actA, aggB, wr3t, wl3t, bl3,
                                                                    nullptr, h3, N_NODES, 0);
    }

    // ---- pool + head ----
    k_pool_sum<<<nblk((long long)N_NODES * 16, B), B, 0, stream>>>(h3, batch, sums, N_NODES);
    k_final<<<1, 64, 0, stream>>>(sums, cnt, Wc, bc, (float*)d_out);
}
